// GraphAutoEncoder_54812372632349
// MI455X (gfx1250) — compile-verified, matrix operand order checked by reading, not run on hardware
//
#include <hip/hip_runtime.h>

#define NODES 100000
#define EMB 32
#define HID 64
#define INF 128

typedef _Float16 v8h  __attribute__((ext_vector_type(8)));
typedef _Float16 v16h __attribute__((ext_vector_type(16)));
typedef float    v8f  __attribute__((ext_vector_type(8)));

// Native FP32 atomic add (global_atomic_add_f32, no CAS loop).
__device__ __forceinline__ void atomic_add_f32(float* p, float v) {
    unsafeAtomicAdd(p, v);
}

// ---------- small utility kernels ----------

__global__ void k_zero_f32(float* __restrict__ p, int n) {
    int i = blockIdx.x * blockDim.x + threadIdx.x;
    if (i < n) p[i] = 0.0f;
}

__global__ void k_deg_count(const int* __restrict__ dst, float* __restrict__ deg, int E) {
    int e = blockIdx.x * blockDim.x + threadIdx.x;
    if (e < E) atomic_add_f32(&deg[dst[e]], 1.0f);
}

__global__ void k_deg_finalize(float* __restrict__ deg, int n) {
    int i = blockIdx.x * blockDim.x + threadIdx.x;
    if (i < n) deg[i] = rsqrtf(deg[i] + 1.0f);   // deg^{-1/2}, incl. self-loop
}

// Wt[o*Kin + i] = (f16) W[i*Nout + o]   (transpose + convert; W is [Kin, Nout])
__global__ void k_wt_f16(const float* __restrict__ W, _Float16* __restrict__ Wt,
                         int Kin, int Nout) {
    int t = blockIdx.x * blockDim.x + threadIdx.x;
    if (t < Kin * Nout) {
        int i = t / Nout, o = t % Nout;
        Wt[o * Kin + i] = (_Float16)W[t];
    }
}

// y16[i] = f16(relu? max(0,x) : x); optionally duplicate the f32 value to `dup`
__global__ void k_cvt_f16(const float* __restrict__ x, _Float16* __restrict__ y,
                          int n, int relu, float* __restrict__ dup) {
    int i = blockIdx.x * blockDim.x + threadIdx.x;
    if (i < n) {
        float v = x[i];
        if (relu) v = v > 0.0f ? v : 0.0f;
        y[i] = (_Float16)v;
        if (dup) dup[i] = v;
    }
}

// ---------- WMMA GEMM: C[M,N] = A[M,K] (f16, row-major) x Bt[N,K]^T (f16) ----------
// One wave computes one 16x16 output tile, iterating K in steps of 32.
// Operand layout per CDNA5 ISA (16-bit A 16x32): lanes 0-15 hold row M=lane,
// K = [0..7] and [16..23]; lanes 16-31 hold K = [8..15] and [24..31].
// B operand mirrors this with columns of B == rows of Bt.
__global__ __launch_bounds__(128)
void k_gemm_wmma(const _Float16* __restrict__ A, const _Float16* __restrict__ Bt,
                 float* __restrict__ C, int M, int N, int K) {
    int wave   = (blockIdx.x * blockDim.x + threadIdx.x) >> 5;
    int lane   = threadIdx.x & 31;
    int tilesN = N >> 4;
    int tilesM = M >> 4;
    if (wave >= tilesM * tilesN) return;          // wave-uniform: EXEC stays all-1s
    int tm = wave / tilesN;
    int tn = wave % tilesN;

    int half = lane >> 4;                          // 0: lanes 0-15, 1: lanes 16-31
    int l    = lane & 15;
    int kb   = half * 8;                           // K sub-offset per ISA layout

    const _Float16* arow = A  + (size_t)(tm * 16 + l) * K + kb;
    const _Float16* brow = Bt + (size_t)(tn * 16 + l) * K + kb;

    v8f c = {};
    for (int kk = 0; kk < K; kk += 32) {
        v8h alo = *(const v8h*)(arow + kk);        // K = kb .. kb+7
        v8h ahi = *(const v8h*)(arow + kk + 16);   // K = kb+16 .. kb+23
        v8h blo = *(const v8h*)(brow + kk);
        v8h bhi = *(const v8h*)(brow + kk + 16);
        v16h av, bv;
#pragma unroll
        for (int i = 0; i < 8; ++i) {
            av[i] = alo[i]; av[i + 8] = ahi[i];
            bv[i] = blo[i]; bv[i + 8] = bhi[i];
        }
        c = __builtin_amdgcn_wmma_f32_16x16x32_f16(
                /*neg_a=*/false, av, /*neg_b=*/false, bv,
                /*c_mod=*/(short)0, c, /*reuse_a=*/false, /*reuse_b=*/false);
    }

    int colo = tn * 16 + l;
#pragma unroll
    for (int r = 0; r < 8; ++r) {                  // VGPR r -> M = r + 8*half
        int rowo = tm * 16 + r + half * 8;
        C[(size_t)rowo * N + colo] = c[r];
    }
}

// ---------- aggregation ----------

// agg[node,f] = h[node,f] * dinv[node]^2 + bias[f]   (self-loop + bias)
template <int F>
__global__ void k_init_agg(float* __restrict__ agg, const float* __restrict__ h,
                           const float* __restrict__ dinv, const float* __restrict__ bias,
                           int n) {
    int t = blockIdx.x * blockDim.x + threadIdx.x;
    if (t < n) {
        int node = t / F;
        int f    = t % F;
        float di = dinv[node];
        agg[t] = h[t] * di * di + bias[f];
    }
}

// agg[dst,f] += h[src,f] * dinv[src] * dinv[dst]   over all edges
template <int F>
__global__ void k_edge_scatter(float* __restrict__ agg, const float* __restrict__ h,
                               const float* __restrict__ dinv,
                               const int* __restrict__ src, const int* __restrict__ dst,
                               int E) {
    long long t = (long long)blockIdx.x * blockDim.x + threadIdx.x;
    if (t >= (long long)E * F) return;
    int e = (int)(t / F);
    int f = (int)(t % F);
    int s = src[e];
    int d = dst[e];
    float v = h[(size_t)s * F + f] * dinv[s] * dinv[d];
    atomic_add_f32(&agg[(size_t)d * F + f], v);
}

// ---------- host side ----------

static inline int blocks_for(long long n, int bs) { return (int)((n + bs - 1) / bs); }

extern "C" void kernel_launch(void* const* d_in, const int* in_sizes, int n_in,
                              void* d_out, int out_size, void* d_ws, size_t ws_size,
                              hipStream_t stream) {
    const int E = in_sizes[0] / 2;
    const int*   src = (const int*)d_in[0];
    const int*   dst = src + E;
    const float* emb = (const float*)d_in[1];
    const float* We1 = (const float*)d_in[2];
    const float* be1 = (const float*)d_in[3];
    const float* We2 = (const float*)d_in[4];
    const float* be2 = (const float*)d_in[5];
    const float* Wd1 = (const float*)d_in[6];
    const float* bd1 = (const float*)d_in[7];
    const float* Wd2 = (const float*)d_in[8];
    const float* bd2 = (const float*)d_in[9];

    float* recon = (float*)d_out;                        // [NODES, 128]
    float* zout  = recon + (size_t)NODES * INF;          // [NODES, 32]

    // workspace carve (256B aligned)
    char*  ws  = (char*)d_ws;
    size_t off = 0;
    auto carve = [&](size_t bytes) -> char* {
        char* p = ws + off;
        off = (off + bytes + 255) & ~(size_t)255;
        return p;
    };
    float*     dinv = (float*)    carve((size_t)NODES * 4);
    _Float16*  a16  = (_Float16*) carve((size_t)NODES * INF * 2);
    float*     h    = (float*)    carve((size_t)NODES * INF * 4);
    float*     agg  = (float*)    carve((size_t)NODES * INF * 4);
    _Float16*  We1t = (_Float16*) carve((size_t)EMB * HID * 2);
    _Float16*  We2t = (_Float16*) carve((size_t)HID * EMB * 2);
    _Float16*  Wd1t = (_Float16*) carve((size_t)EMB * HID * 2);
    _Float16*  Wd2t = (_Float16*) carve((size_t)HID * INF * 2);

    const int BS = 256;

    // degree / normalization (shared by all layers; depends only on dst)
    k_zero_f32<<<blocks_for(NODES, BS), BS, 0, stream>>>(dinv, NODES);
    k_deg_count<<<blocks_for(E, BS), BS, 0, stream>>>(dst, dinv, E);
    k_deg_finalize<<<blocks_for(NODES, BS), BS, 0, stream>>>(dinv, NODES);

    // weight transpose + f16 convert
    k_wt_f16<<<blocks_for(EMB * HID, BS), BS, 0, stream>>>(We1, We1t, EMB, HID);
    k_wt_f16<<<blocks_for(HID * EMB, BS), BS, 0, stream>>>(We2, We2t, HID, EMB);
    k_wt_f16<<<blocks_for(EMB * HID, BS), BS, 0, stream>>>(Wd1, Wd1t, EMB, HID);
    k_wt_f16<<<blocks_for(HID * INF, BS), BS, 0, stream>>>(Wd2, Wd2t, HID, INF);

    auto gemm = [&](const _Float16* A, const _Float16* Bt, float* C, int N, int K) {
        long long waves = (long long)(NODES / 16) * (N / 16);
        k_gemm_wmma<<<blocks_for(waves * 32, 128), 128, 0, stream>>>(A, Bt, C, NODES, N, K);
    };

    // ---- Layer 1 (encoder): emb[.,32] -> h[.,64], relu ----
    k_cvt_f16<<<blocks_for((long long)NODES * EMB, BS), BS, 0, stream>>>(
        emb, a16, NODES * EMB, 0, nullptr);
    gemm(a16, We1t, h, HID, EMB);
    k_init_agg<HID><<<blocks_for((long long)NODES * HID, BS), BS, 0, stream>>>(
        agg, h, dinv, be1, NODES * HID);
    k_edge_scatter<HID><<<blocks_for((long long)E * HID, BS), BS, 0, stream>>>(
        agg, h, dinv, src, dst, E);
    k_cvt_f16<<<blocks_for((long long)NODES * HID, BS), BS, 0, stream>>>(
        agg, a16, NODES * HID, 1, nullptr);

    // ---- Layer 2 (encoder): [.,64] -> z[.,32], relu; dup z into d_out ----
    gemm(a16, We2t, h, EMB, HID);
    k_init_agg<EMB><<<blocks_for((long long)NODES * EMB, BS), BS, 0, stream>>>(
        agg, h, dinv, be2, NODES * EMB);
    k_edge_scatter<EMB><<<blocks_for((long long)E * EMB, BS), BS, 0, stream>>>(
        agg, h, dinv, src, dst, E);
    k_cvt_f16<<<blocks_for((long long)NODES * EMB, BS), BS, 0, stream>>>(
        agg, a16, NODES * EMB, 1, zout);

    // ---- Layer 3 (decoder): z[.,32] -> y[.,64], relu ----
    gemm(a16, Wd1t, h, HID, EMB);
    k_init_agg<HID><<<blocks_for((long long)NODES * HID, BS), BS, 0, stream>>>(
        agg, h, dinv, bd1, NODES * HID);
    k_edge_scatter<HID><<<blocks_for((long long)E * HID, BS), BS, 0, stream>>>(
        agg, h, dinv, src, dst, E);
    k_cvt_f16<<<blocks_for((long long)NODES * HID, BS), BS, 0, stream>>>(
        agg, a16, NODES * HID, 1, nullptr);

    // ---- Layer 4 (decoder): y[.,64] -> recon_x[.,128], no relu, straight to d_out ----
    gemm(a16, Wd2t, h, INF, HID);
    k_init_agg<INF><<<blocks_for((long long)NODES * INF, BS), BS, 0, stream>>>(
        recon, h, dinv, bd2, NODES * INF);
    k_edge_scatter<INF><<<blocks_for((long long)E * INF, BS), BS, 0, stream>>>(
        recon, h, dinv, src, dst, E);
}
